// NFLayer_69346541962024
// MI455X (gfx1250) — compile-verified, matrix operand order checked by reading, not run on hardware
//
#include <hip/hip_runtime.h>
#include <hip/hip_bf16.h>

// ---------------- problem constants ----------------
static constexpr int N_NODES = 20000;
static constexpr int N_EDGES = 160000;
static constexpr int D_IN    = 512;
static constexpr int D_OUT   = 512;
static constexpr int KCAT    = 1024;     // concatenated [h | x] K dimension
static constexpr int MAXDEG  = 10;
static constexpr int NBUCK   = MAXDEG + 1;
static constexpr float BN_EPS = 1e-5f;

static constexpr int MTILE   = 32;                       // nodes per GEMM block
static constexpr int MAXT    = N_NODES / MTILE + NBUCK;  // upper bound on tiles = 636
static constexpr int A_ROW_BYTES = KCAT * 2;             // 2048
static constexpr int A_ROW_PITCH = A_ROW_BYTES + 16;     // 2064: bank-conflict pad

// ---------------- workspace layout (bytes) ----------------
static constexpr size_t OFF_H     = 0;                                   // N*512 f32 (aliased as out_pre)
static constexpr size_t OFF_ABF   = OFF_H     + (size_t)N_NODES*512*4;   // N*1024 bf16
static constexpr size_t OFF_W     = OFF_ABF   + (size_t)N_NODES*KCAT*2;  // 11*512*1024 bf16 (transposed)
static constexpr size_t OFF_DEG   = OFF_W     + (size_t)NBUCK*D_OUT*KCAT*2;
static constexpr size_t OFF_PERM  = OFF_DEG   + (size_t)N_NODES*4;
static constexpr size_t OFF_CNT   = OFF_PERM  + (size_t)N_NODES*4;       // 16 ints
static constexpr size_t OFF_CUR   = OFF_CNT   + 64;                      // 16 ints
static constexpr size_t OFF_META  = OFF_CUR   + 64;                      // numTiles
static constexpr size_t OFF_TDEG  = OFF_META  + 64;
static constexpr size_t OFF_TSTART= OFF_TDEG  + (size_t)MAXT*4;
static constexpr size_t OFF_TCNT  = OFF_TSTART+ (size_t)MAXT*4;
static constexpr size_t OFF_SUM   = (OFF_TCNT + (size_t)MAXT*4 + 255) & ~size_t(255);
static constexpr size_t OFF_SQ    = OFF_SUM   + 512*4;
static constexpr size_t OFF_SCALE = OFF_SQ    + 512*4;
static constexpr size_t OFF_SHIFT = OFF_SCALE + 512*4;

// ---------------- types ----------------
typedef __attribute__((ext_vector_type(16))) __bf16 v16bf;
typedef __attribute__((ext_vector_type(8)))  float  v8f;

union FragU { v16bf v; uint4 q[2]; };

__device__ __forceinline__ unsigned short f32_to_bf16(float f) {
    unsigned int u = __float_as_uint(f);
    unsigned int r = u + 0x7FFFu + ((u >> 16) & 1u);   // round-to-nearest-even
    if ((u & 0x7F800000u) == 0x7F800000u) r = u;       // inf/nan passthrough
    return (unsigned short)(r >> 16);
}

// monotonic order-preserving f32 <-> u32 map (for atomic max)
__device__ __forceinline__ unsigned int enc_f32(float f) {
    unsigned int u = __float_as_uint(f);
    return (u & 0x80000000u) ? ~u : (u | 0x80000000u);
}
__device__ __forceinline__ float dec_f32(unsigned int k) {
    unsigned int u = (k & 0x80000000u) ? (k ^ 0x80000000u) : ~k;
    return __uint_as_float(u);
}

// ---------------- kernels ----------------

// in-degree (order-independent int atomics)
__global__ void k_deg(const int* __restrict__ dst, int* __restrict__ deg) {
    int e = blockIdx.x * blockDim.x + threadIdx.x;
    if (e < N_EDGES) atomicAdd(&deg[dst[e]], 1);
}

// h[dst] += feats[src] : one thread per (edge, 4 channels)
__global__ void k_edge_sum(const float* __restrict__ feats, const int* __restrict__ src,
                           const int* __restrict__ dst, float* __restrict__ h) {
    long long tid = (long long)blockIdx.x * blockDim.x + threadIdx.x;
    int e = (int)(tid >> 7);
    if (e >= N_EDGES) return;
    int part = (int)(tid & 127);
    int c0 = part * 4;
    const float4 v = *(const float4*)(feats + (size_t)src[e] * D_IN + c0);
    float* hp = h + (size_t)dst[e] * D_IN + c0;
    atomicAdd(hp + 0, v.x);
    atomicAdd(hp + 1, v.y);
    atomicAdd(hp + 2, v.z);
    atomicAdd(hp + 3, v.w);
}

__global__ void k_count(const int* __restrict__ deg, int* __restrict__ cnt) {
    int n = blockIdx.x * blockDim.x + threadIdx.x;
    if (n < N_NODES) {
        int d = deg[n]; if (d > MAXDEG) d = MAXDEG;
        atomicAdd(&cnt[d], 1);
    }
}

// single-thread prep: prefix sums, cursors, per-degree tile table
__global__ void k_prep(const int* __restrict__ cnt, int* __restrict__ cur,
                       int* __restrict__ numTiles, int* __restrict__ tdeg,
                       int* __restrict__ tstart, int* __restrict__ tcnt) {
    if (threadIdx.x != 0 || blockIdx.x != 0) return;
    int bo[NBUCK + 1];
    bo[0] = 0;
    for (int d = 0; d < NBUCK; ++d) bo[d + 1] = bo[d] + cnt[d];
    int t = 0;
    for (int d = 0; d < NBUCK; ++d) {
        cur[d] = bo[d];
        int c = cnt[d];
        for (int i = 0; i < c; i += MTILE) {
            tdeg[t]   = d;
            tstart[t] = bo[d] + i;
            tcnt[t]   = (c - i < MTILE) ? (c - i) : MTILE;
            ++t;
        }
    }
    *numTiles = t;
}

__global__ void k_scatter(const int* __restrict__ deg, int* __restrict__ cur,
                          int* __restrict__ perm) {
    int n = blockIdx.x * blockDim.x + threadIdx.x;
    if (n < N_NODES) {
        int d = deg[n]; if (d > MAXDEG) d = MAXDEG;
        int pos = atomicAdd(&cur[d], 1);
        perm[pos] = n;
    }
}

// pack A = [h | x] -> bf16 [N, 1024]; 1 thread = 2 consecutive k (one dword)
__global__ void k_pack_a(const float* __restrict__ h, const float* __restrict__ feats,
                         unsigned int* __restrict__ abf) {
    long long tid = (long long)blockIdx.x * blockDim.x + threadIdx.x;
    if (tid >= (long long)N_NODES * 512) return;
    int n = (int)(tid / 512);
    int p = (int)(tid % 512);
    int k0 = 2 * p;
    const float* srcbuf = (k0 < D_IN) ? (h + (size_t)n * D_IN + k0)
                                      : (feats + (size_t)n * D_IN + (k0 - D_IN));
    unsigned int lo = f32_to_bf16(srcbuf[0]);
    unsigned int hi = f32_to_bf16(srcbuf[1]);
    abf[(size_t)n * (KCAT / 2) + p] = (hi << 16) | lo;
}

// pack W^T: wt[d][o][k] = (k<512 ? Wl[d][k][o] : Wr[d][k-512][o]) as bf16
__global__ void k_pack_w(const float* __restrict__ Wl, const float* __restrict__ Wr,
                         unsigned int* __restrict__ wt) {
    long long tid = (long long)blockIdx.x * blockDim.x + threadIdx.x;
    if (tid >= (long long)NBUCK * D_OUT * 512) return;
    int p = (int)(tid % 512);
    int o = (int)((tid / 512) % D_OUT);
    int d = (int)(tid / (512 * D_OUT));
    int k0 = 2 * p;
    float a, b;
    if (k0 < D_IN) {
        const float* base = Wl + (size_t)d * D_IN * D_OUT + (size_t)k0 * D_OUT + o;
        a = base[0]; b = base[D_OUT];
    } else {
        const float* base = Wr + (size_t)d * D_IN * D_OUT + (size_t)(k0 - D_IN) * D_OUT + o;
        a = base[0]; b = base[D_OUT];
    }
    unsigned int lo = f32_to_bf16(a);
    unsigned int hi = f32_to_bf16(b);
    wt[((size_t)d * D_OUT + o) * (KCAT / 2) + p] = (hi << 16) | lo;
}

// WMMA GEMM: out_pre[node, :] = [h|x] @ [Wl;Wr]^cat + bl   (per-degree tiles)
// M-tile = 32 nodes (2 WMMA M-subtiles -> each B fragment feeds 2 WMMAs).
// Both A (LDS) and B (global/L2) fragments are register double-buffered so all
// loads for step ks+1 retire behind the 8 WMMAs of step ks.
__global__ void __launch_bounds__(256) k_gemm(
    const unsigned short* __restrict__ abf,   // [N, 1024] bf16
    const unsigned short* __restrict__ wt,    // [11, 512, 1024] bf16 (o-major)
    const float* __restrict__ bl,             // [11, 512]
    const int* __restrict__ tdeg, const int* __restrict__ tstart,
    const int* __restrict__ tcnt, const int* __restrict__ numTiles,
    const int* __restrict__ perm,
    float* __restrict__ outp)                 // [N, 512]
{
    if ((int)blockIdx.x >= *numTiles) return;
    __shared__ int nodeIds[MTILE];
    __shared__ unsigned char smA[MTILE * A_ROW_PITCH];

    const int tid = threadIdx.x;
    const int t   = blockIdx.x;
    const int d   = tdeg[t];
    const int st  = tstart[t];
    const int cn  = tcnt[t];

    if (tid < MTILE) nodeIds[tid] = (tid < cn) ? perm[st + tid] : -1;
    __syncthreads();

    // stage A tile into LDS: 32 rows x 2048B, 8 threads per row, b128 chunks
    {
        const int r = tid >> 3, c8 = tid & 7;
        const int node = nodeIds[r];
        uint4* drow = (uint4*)(smA + (size_t)r * A_ROW_PITCH);
        const uint4* srow = (node >= 0) ? (const uint4*)(abf + (size_t)node * KCAT) : nullptr;
        uint4 z; z.x = z.y = z.z = z.w = 0u;
#pragma unroll
        for (int j = 0; j < 16; ++j) {
            int c = c8 + j * 8;            // 128 chunks of 16B per row
            drow[c] = srow ? srow[c] : z;
        }
    }
    __syncthreads();

    const int w    = tid >> 5;
    const int lane = tid & 31;
    const int l16  = lane & 15;
    const int kh   = lane >> 4;            // K-half select per WMMA bf16 layout
    const unsigned char* wbase = (const unsigned char*)(wt + (size_t)d * D_OUT * KCAT);
    const int colb = w * 64;
    // per-lane base of the B column this lane supplies (ks and nt become immediates)
    const unsigned char* bcol = wbase + (size_t)(colb + l16) * A_ROW_BYTES + kh * 16;
    const unsigned char* arow0 = smA + (size_t)l16 * A_ROW_PITCH + kh * 16;
    const unsigned char* arow1 = arow0 + 16 * A_ROW_PITCH;

    v8f acc[2][4];
#pragma unroll
    for (int ms = 0; ms < 2; ++ms)
#pragma unroll
        for (int nt = 0; nt < 4; ++nt)
            acc[ms][nt] = v8f{0.f,0.f,0.f,0.f,0.f,0.f,0.f,0.f};

    FragU b[2][4];
    FragU a[2][2];
    // prologue: fetch A and B fragments for ks = 0
#pragma unroll
    for (int nt = 0; nt < 4; ++nt) {
        const unsigned char* bp = bcol + (size_t)nt * 16 * A_ROW_BYTES;
        b[0][nt].q[0] = *(const uint4*)bp;
        b[0][nt].q[1] = *(const uint4*)(bp + 32);
    }
    a[0][0].q[0] = *(const uint4*)(arow0);
    a[0][0].q[1] = *(const uint4*)(arow0 + 32);
    a[0][1].q[0] = *(const uint4*)(arow1);
    a[0][1].q[1] = *(const uint4*)(arow1 + 32);

#pragma unroll 2
    for (int ks = 0; ks < KCAT / 32; ++ks) {
        const int cb = ks & 1, nb = cb ^ 1;
        const int ksn = (ks + 1) & (KCAT / 32 - 1);   // wraps: last prefetch is harmless
        // prefetch next B fragments (global/L2) — overlaps with WMMAs below
#pragma unroll
        for (int nt = 0; nt < 4; ++nt) {
            const unsigned char* bp = bcol + (size_t)nt * 16 * A_ROW_BYTES + ksn * 64;
            b[nb][nt].q[0] = *(const uint4*)bp;
            b[nb][nt].q[1] = *(const uint4*)(bp + 32);
        }
        // prefetch next A fragments (LDS) — overlaps with WMMAs below
        a[nb][0].q[0] = *(const uint4*)(arow0 + ksn * 64);
        a[nb][0].q[1] = *(const uint4*)(arow0 + ksn * 64 + 32);
        a[nb][1].q[0] = *(const uint4*)(arow1 + ksn * 64);
        a[nb][1].q[1] = *(const uint4*)(arow1 + ksn * 64 + 32);
#pragma unroll
        for (int nt = 0; nt < 4; ++nt) {
            acc[0][nt] = __builtin_amdgcn_wmma_f32_16x16x32_bf16(
                false, a[cb][0].v, false, b[cb][nt].v, (short)0, acc[0][nt], false, false);
            acc[1][nt] = __builtin_amdgcn_wmma_f32_16x16x32_bf16(
                false, a[cb][1].v, false, b[cb][nt].v, (short)0, acc[1][nt], false, false);
        }
    }

    // epilogue: C layout -> lane l16 = col-in-tile, element j -> row ms*16 + j + kh*8
#pragma unroll
    for (int ms = 0; ms < 2; ++ms) {
#pragma unroll
        for (int nt = 0; nt < 4; ++nt) {
            const int col  = colb + nt * 16 + l16;
            const float bias = bl[d * D_OUT + col];
#pragma unroll
            for (int j = 0; j < 8; ++j) {
                const int m = ms * 16 + j + kh * 8;
                const int node = nodeIds[m];
                if (node >= 0) outp[(size_t)node * D_OUT + col] = acc[ms][nt][j] + bias;
            }
        }
    }
}

// BN partial stats: each block covers 250 rows; thread owns channels c and c+256
__global__ void k_bn_partial(const float* __restrict__ outp,
                             float* __restrict__ sum, float* __restrict__ sq) {
    const int c0 = threadIdx.x;
    const int r0 = blockIdx.x * 250;
    float s0 = 0.f, s1 = 0.f, q0 = 0.f, q1 = 0.f;
    for (int i = 0; i < 250; ++i) {
        const float* row = outp + (size_t)(r0 + i) * D_OUT;
        float a = row[c0], b = row[c0 + 256];
        s0 += a; q0 += a * a;
        s1 += b; q1 += b * b;
    }
    atomicAdd(&sum[c0], s0);       atomicAdd(&sq[c0], q0);
    atomicAdd(&sum[c0 + 256], s1); atomicAdd(&sq[c0 + 256], q1);
}

__global__ void k_bn_final(const float* __restrict__ sum, const float* __restrict__ sq,
                           const float* __restrict__ gamma, const float* __restrict__ beta,
                           float* __restrict__ scale, float* __restrict__ shift) {
    int c = blockIdx.x * blockDim.x + threadIdx.x;
    if (c < D_OUT) {
        float mean = sum[c] * (1.0f / N_NODES);
        float var  = sq[c] * (1.0f / N_NODES) - mean * mean;
        float rstd = rsqrtf(var + BN_EPS);
        float sc = gamma[c] * rstd;
        scale[c] = sc;
        shift[c] = beta[c] - mean * sc;
    }
}

// init output with encoded self-loop value (covers segment_max identity)
__global__ void k_init_out(const float* __restrict__ outp, const float* __restrict__ scale,
                           const float* __restrict__ shift, unsigned int* __restrict__ outk) {
    long long tid = (long long)blockIdx.x * blockDim.x + threadIdx.x;
    if (tid >= (long long)N_NODES * 128) return;
    int n = (int)(tid >> 7);
    int c0 = (int)(tid & 127) * 4;
    const float4 v = *(const float4*)(outp + (size_t)n * D_OUT + c0);
    uint4 k;
    k.x = enc_f32(v.x * scale[c0 + 0] + shift[c0 + 0]);
    k.y = enc_f32(v.y * scale[c0 + 1] + shift[c0 + 1]);
    k.z = enc_f32(v.z * scale[c0 + 2] + shift[c0 + 2]);
    k.w = enc_f32(v.w * scale[c0 + 3] + shift[c0 + 3]);
    *(uint4*)(outk + (size_t)n * D_OUT + c0) = k;
}

// per-edge segment max (order-independent uint atomics)
__global__ void k_edge_max(const float* __restrict__ outp, const float* __restrict__ scale,
                           const float* __restrict__ shift, const int* __restrict__ src,
                           const int* __restrict__ dst, unsigned int* __restrict__ outk) {
    long long tid = (long long)blockIdx.x * blockDim.x + threadIdx.x;
    int e = (int)(tid >> 7);
    if (e >= N_EDGES) return;
    int c0 = (int)(tid & 127) * 4;
    const float4 v = *(const float4*)(outp + (size_t)src[e] * D_OUT + c0);
    unsigned int* op = outk + (size_t)dst[e] * D_OUT + c0;
    atomicMax(op + 0, enc_f32(v.x * scale[c0 + 0] + shift[c0 + 0]));
    atomicMax(op + 1, enc_f32(v.y * scale[c0 + 1] + shift[c0 + 1]));
    atomicMax(op + 2, enc_f32(v.z * scale[c0 + 2] + shift[c0 + 2]));
    atomicMax(op + 3, enc_f32(v.w * scale[c0 + 3] + shift[c0 + 3]));
}

__global__ void k_decode(unsigned int* __restrict__ outk) {
    long long tid = (long long)blockIdx.x * blockDim.x + threadIdx.x;
    if (tid >= (long long)N_NODES * 128) return;
    uint4* p = (uint4*)outk + tid;
    uint4 k = *p;
    float4 f;
    f.x = dec_f32(k.x); f.y = dec_f32(k.y); f.z = dec_f32(k.z); f.w = dec_f32(k.w);
    *(float4*)p = f;
}

// ---------------- launch ----------------
extern "C" void kernel_launch(void* const* d_in, const int* in_sizes, int n_in,
                              void* d_out, int out_size, void* d_ws, size_t ws_size,
                              hipStream_t stream) {
    const float* feats = (const float*)d_in[0];
    const int*   src   = (const int*)  d_in[1];
    const int*   dst   = (const int*)  d_in[2];
    const float* Wl    = (const float*)d_in[3];
    const float* Wr    = (const float*)d_in[4];
    const float* bl    = (const float*)d_in[5];
    const float* gamma = (const float*)d_in[6];
    const float* beta  = (const float*)d_in[7];

    char* ws = (char*)d_ws;
    float*          h      = (float*)(ws + OFF_H);
    float*          outp   = (float*)(ws + OFF_H);      // aliased: h dead after packing
    unsigned short* abf    = (unsigned short*)(ws + OFF_ABF);
    unsigned short* wt     = (unsigned short*)(ws + OFF_W);
    int*            deg    = (int*)(ws + OFF_DEG);
    int*            perm   = (int*)(ws + OFF_PERM);
    int*            cnt    = (int*)(ws + OFF_CNT);
    int*            cur    = (int*)(ws + OFF_CUR);
    int*            nTiles = (int*)(ws + OFF_META);
    int*            tdeg   = (int*)(ws + OFF_TDEG);
    int*            tstart = (int*)(ws + OFF_TSTART);
    int*            tcnt   = (int*)(ws + OFF_TCNT);
    float*          sum    = (float*)(ws + OFF_SUM);
    float*          sq     = (float*)(ws + OFF_SQ);
    float*          scale  = (float*)(ws + OFF_SCALE);
    float*          shift  = (float*)(ws + OFF_SHIFT);
    unsigned int*   outk   = (unsigned int*)d_out;

    // zero accumulators (graph-capture-safe async memsets)
    hipMemsetAsync(h, 0, (size_t)N_NODES * D_IN * 4, stream);
    hipMemsetAsync(deg, 0, (size_t)N_NODES * 4, stream);
    hipMemsetAsync(cnt, 0, 64, stream);
    hipMemsetAsync(sum, 0, 2 * 512 * 4, stream);

    const int B = 256;
    // 1) neighbor sum + degrees
    k_deg<<<(N_EDGES + B - 1) / B, B, 0, stream>>>(dst, deg);
    k_edge_sum<<<(int)(((long long)N_EDGES * 128 + B - 1) / B), B, 0, stream>>>(feats, src, dst, h);
    // 2) degree bucketing + tile table
    k_count<<<(N_NODES + B - 1) / B, B, 0, stream>>>(deg, cnt);
    k_prep<<<1, 1, 0, stream>>>(cnt, cur, nTiles, tdeg, tstart, tcnt);
    k_scatter<<<(N_NODES + B - 1) / B, B, 0, stream>>>(deg, cur, perm);
    // 3) bf16 packing
    k_pack_a<<<(int)(((long long)N_NODES * 512 + B - 1) / B), B, 0, stream>>>(h, feats, (unsigned int*)abf);
    k_pack_w<<<(int)(((long long)NBUCK * D_OUT * 512 + B - 1) / B), B, 0, stream>>>(Wl, Wr, (unsigned int*)wt);
    // 4) WMMA GEMM (out_pre aliases h; abf already holds h in bf16)
    k_gemm<<<MAXT, B, 0, stream>>>(abf, wt, bl, tdeg, tstart, tcnt, nTiles, perm, outp);
    // 5) batchnorm stats -> scale/shift
    k_bn_partial<<<N_NODES / 250, B, 0, stream>>>(outp, sum, sq);
    k_bn_final<<<2, B, 0, stream>>>(sum, sq, gamma, beta, scale, shift);
    // 6) segment max with self loops (encoded monotonic uint atomics)
    k_init_out<<<(int)(((long long)N_NODES * 128 + B - 1) / B), B, 0, stream>>>(outp, scale, shift, outk);
    k_edge_max<<<(int)(((long long)N_EDGES * 128 + B - 1) / B), B, 0, stream>>>(outp, scale, shift, src, dst, outk);
    k_decode<<<(int)(((long long)N_NODES * 128 + B - 1) / B), B, 0, stream>>>(outk);
}
